// TransformerBlock_72696616452196
// MI455X (gfx1250) — matrix operand order repaired and checked by reading, not compiled, not run on hardware
//
#include <hip/hip_runtime.h>

typedef __attribute__((ext_vector_type(16))) __bf16 v16bf;
typedef __attribute__((ext_vector_type(8)))  float  v8f;

#define USE_ASYNC_LDS 1

static __device__ __forceinline__ unsigned short f2bf(float f) {
    unsigned int u = __float_as_uint(f);
    u += 0x7FFFu + ((u >> 16) & 1u);   // round-to-nearest-even
    return (unsigned short)(u >> 16);
}

union FragB { v16bf v; unsigned int u[8]; unsigned short s[16]; };

#if USE_ASYNC_LDS
// GLOBAL_LOAD_ASYNC_TO_LDS_B128: per-lane 16B global -> LDS, tracked by ASYNCcnt.
// vdst = LDS byte address (low 32 bits of flat LDS address), vaddr = 64-bit global.
static __device__ __forceinline__ void async_copy16(void* lds, const void* g) {
    unsigned int l = (unsigned int)(size_t)lds;
    unsigned long long ga = (unsigned long long)(size_t)g;
    asm volatile("global_load_async_to_lds_b128 %0, %1, off" :: "v"(l), "v"(ga) : "memory");
}
static __device__ __forceinline__ void async_wait0() {
    asm volatile("s_wait_asynccnt 0" ::: "memory");
}
#endif

// ---------------- fp32 -> bf16 conversion ----------------
__global__ void k_f32_to_bf16(const float* __restrict__ in,
                              unsigned short* __restrict__ out, int n) {
    int i = blockIdx.x * blockDim.x + threadIdx.x;
    int stride = gridDim.x * blockDim.x;
    for (; i < n; i += stride) out[i] = f2bf(in[i]);
}

// ---------------- bf16 WMMA GEMM: C = A[M,K] * B[K,N] + bias ----------------
// MODE 0: bf16 row-major out (RELU optional)
// MODE 1: f32 row-major out
// MODE 2: bf16 head-split out [B,H,S,64] (QKV projection)
template <int MODE, int RELU>
__global__ __launch_bounds__(256)
void k_gemm_bf16(const unsigned short* __restrict__ A,
                 const unsigned short* __restrict__ Bm,
                 const float* __restrict__ bias,
                 void* __restrict__ out,
                 int M, int N, int K)
{
    // double-buffered tiles; A stride 40 halves (80B) keeps 16B chunks aligned
    __shared__ __align__(16) unsigned short As[2][128][40];
    __shared__ unsigned short Bst[2][128][34];   // B transposed: [n][k]

    const int tid  = threadIdx.x;
    const int lane = tid & 31;
    const int wave = tid >> 5;     // 0..7
    const int wm   = wave >> 2;    // 0..1  (M direction)
    const int wn   = wave & 3;     // 0..3  (N direction)
    const int l15  = lane & 15;
    const int h2   = lane >> 4;

    const int n0 = blockIdx.x * 128;
    const int m0 = blockIdx.y * 128;

    v8f acc[4][2];
    for (int mi = 0; mi < 4; ++mi)
        for (int ni = 0; ni < 2; ++ni)
            for (int e = 0; e < 8; ++e) acc[mi][ni][e] = 0.0f;

    const int arow = tid >> 1, aseg = tid & 1;   // A tile: 128 rows x (2 x 16) cols
    const int brow = tid >> 3, bseg = tid & 7;   // B tile: 32 rows x (8 x 16) cols

    auto stage = [&](int k0, int buf) {
        const unsigned short* srcA = A + (size_t)(m0 + arow) * K + k0 + aseg * 16;
#if USE_ASYNC_LDS
        async_copy16(&As[buf][arow][aseg * 16],     srcA);
        async_copy16(&As[buf][arow][aseg * 16 + 8], srcA + 8);
#else
        unsigned int* dst = (unsigned int*)&As[buf][arow][aseg * 16];
        #pragma unroll
        for (int j = 0; j < 8; ++j) dst[j] = ((const unsigned int*)srcA)[j];
#endif
        const unsigned short* srcB = Bm + (size_t)(k0 + brow) * N + n0 + bseg * 16;
        #pragma unroll
        for (int j = 0; j < 16; ++j) Bst[buf][bseg * 16 + j][brow] = srcB[j];
    };

    auto compute = [&](int buf) {
        FragB bf[2];
        #pragma unroll
        for (int ni = 0; ni < 2; ++ni) {
            const int col = wn * 32 + ni * 16 + l15;
            const unsigned int* p = (const unsigned int*)&Bst[buf][col][16 * h2];
            #pragma unroll
            for (int j = 0; j < 8; ++j) bf[ni].u[j] = p[j];
        }
        #pragma unroll
        for (int mi = 0; mi < 4; ++mi) {
            FragB af;
            const int row = wm * 64 + mi * 16 + l15;
            #pragma unroll
            for (int j = 0; j < 8; ++j) {
                const int kk = 2 * j + 16 * h2;
                af.u[j] = *(const unsigned int*)&As[buf][row][kk];
            }
            #pragma unroll
            for (int ni = 0; ni < 2; ++ni) {
                acc[mi][ni] = __builtin_amdgcn_wmma_f32_16x16x32_bf16(
                    false, af.v, false, bf[ni].v, (short)0, acc[mi][ni], false, false);
            }
        }
    };

    const int niter = K >> 5;
    stage(0, 0);
#if USE_ASYNC_LDS
    async_wait0();
#endif
    __syncthreads();
    for (int it = 0; it < niter; ++it) {
        const int cur = it & 1;
        if (it + 1 < niter) stage((it + 1) << 5, cur ^ 1);
        compute(cur);
#if USE_ASYNC_LDS
        async_wait0();
#endif
        __syncthreads();
    }

    // epilogue (compile-time MODE -> single path, no accumulator spill)
    for (int mi = 0; mi < 4; ++mi) {
        for (int ni = 0; ni < 2; ++ni) {
            const int gcol = n0 + wn * 32 + ni * 16 + l15;
            const float bia = bias[gcol];
            #pragma unroll
            for (int r = 0; r < 8; ++r) {
                const int grow = m0 + wm * 64 + mi * 16 + r + 8 * h2;
                float val = acc[mi][ni][r] + bia;
                if (RELU) val = fmaxf(val, 0.0f);
                if (MODE == 1) {
                    ((float*)out)[(size_t)grow * N + gcol] = val;
                } else if (MODE == 0) {
                    ((unsigned short*)out)[(size_t)grow * N + gcol] = f2bf(val);
                } else {
                    const int bb = grow >> 11, ss = grow & 2047;   // S = 2048
                    const int hh = gcol >> 6,  nn = gcol & 63;     // hd = 64
                    ((unsigned short*)out)[((((size_t)bb * 12 + hh) << 11) + ss) * 64 + nn] =
                        f2bf(val);
                }
            }
        }
    }
}

// ---------------- flash attention: one wave per 16-query tile ----------------
// Q,K,V: bf16 [B*H][2048][64]; output: bf16 [B*2048][768] (heads concatenated)
__global__ __launch_bounds__(128)
void k_attention(const unsigned short* __restrict__ Q,
                 const unsigned short* __restrict__ Km,
                 const unsigned short* __restrict__ Vm,
                 unsigned short* __restrict__ Obf)
{
    __shared__ unsigned short Vt[4][64][34];   // per-wave V^T block: [hd][key-in-block]
    __shared__ unsigned short Pt[4][16][34];   // per-wave P tile (A-layout staging)
    __shared__ float Mrow[4][16];
    __shared__ float Lrow[4][16];

    const int lane = threadIdx.x & 31;
    const int w    = threadIdx.x >> 5;           // wave in block
    const int gw   = blockIdx.x * 4 + w;         // global wave id, 3072 total
    const int qt   = gw & 127;                   // 128 q-tiles of 16 rows
    const int bh   = gw >> 7;                    // 0..23 (b*12+h)
    const int b    = bh / 12, h = bh % 12;
    const int q0   = qt * 16;
    const int l15  = lane & 15;
    const int h2   = lane >> 4;

    const size_t head = (size_t)bh * 2048 * 64;
    const unsigned short* Qh = Q + head;
    const unsigned short* Kh = Km + head;
    const unsigned short* Vh = Vm + head;

    if (lane < 16) { Mrow[w][lane] = -3.0e38f; Lrow[w][lane] = 0.0f; }

    // Q fragments (hd=64 -> two K=32 chunks), gathered straight in A-layout
    FragB aQ[2];
    #pragma unroll
    for (int c = 0; c < 2; ++c) {
        const unsigned short* qrow = Qh + (size_t)(q0 + l15) * 64;
        #pragma unroll
        for (int j = 0; j < 8; ++j) {
            const int kk = 2 * j + 16 * h2 + 32 * c;
            aQ[c].u[j] = *(const unsigned int*)(qrow + kk);
        }
    }

    v8f O[4];
    for (int t = 0; t < 4; ++t)
        for (int e = 0; e < 8; ++e) O[t][e] = 0.0f;

    for (int kb = 0; kb < 2048; kb += 32) {
        // stage V block transposed: Vt[n][k] = V[kb+k][n]; lane owns key kb+lane
        {
            const unsigned int* vr = (const unsigned int*)(Vh + (size_t)(kb + lane) * 64);
            #pragma unroll
            for (int j = 0; j < 32; ++j) {
                unsigned int pv = vr[j];
                Vt[w][2 * j][lane]     = (unsigned short)(pv & 0xFFFFu);
                Vt[w][2 * j + 1][lane] = (unsigned short)(pv >> 16);
            }
        }
        // scores for two 16-key tiles: S = Q * K^T  (B[k][n]=K[key n][k], contiguous)
        v8f sc[2];
        #pragma unroll
        for (int t = 0; t < 2; ++t) {
            for (int e = 0; e < 8; ++e) sc[t][e] = 0.0f;
            #pragma unroll
            for (int c = 0; c < 2; ++c) {
                FragB bK;
                const unsigned int* kp = (const unsigned int*)
                    (Kh + (size_t)(kb + t * 16 + l15) * 64 + 32 * c + 16 * h2);
                #pragma unroll
                for (int j = 0; j < 8; ++j) bK.u[j] = kp[j];
                sc[t] = __builtin_amdgcn_wmma_f32_16x16x32_bf16(
                    false, aQ[c].v, false, bK.v, (short)0, sc[t], false, false);
            }
        }
        __syncthreads();

        // online softmax, row-wise over D-layout (row r+8*h2, col = lane&15)
        #pragma unroll
        for (int r = 0; r < 8; ++r) {
            const int row = r + 8 * h2;
            float s0 = sc[0][r] * 0.125f;   // 1/sqrt(64)
            float s1 = sc[1][r] * 0.125f;
            float mt = fmaxf(s0, s1);
            for (int m = 1; m < 16; m <<= 1) mt = fmaxf(mt, __shfl_xor(mt, m, 32));
            float Mold = Mrow[w][row];
            float Mnew = fmaxf(Mold, mt);
            float p0 = __expf(s0 - Mnew);
            float p1 = __expf(s1 - Mnew);
            float lt = p0 + p1;
            for (int m = 1; m < 16; m <<= 1) lt += __shfl_xor(lt, m, 32);
            float corr = __expf(Mold - Mnew);
            if (l15 == 0) { Mrow[w][row] = Mnew; Lrow[w][row] = Lrow[w][row] * corr + lt; }
            Pt[w][row][l15]      = f2bf(p0);
            Pt[w][row][16 + l15] = f2bf(p1);
            #pragma unroll
            for (int t = 0; t < 4; ++t) O[t][r] *= corr;
        }
        __syncthreads();

        // P fragment (A-layout) from LDS
        FragB pf;
        #pragma unroll
        for (int j = 0; j < 8; ++j) {
            const int kk = 2 * j + 16 * h2;
            pf.u[j] = *(const unsigned int*)&Pt[w][l15][kk];
        }
        // O += P(16x32) * V(32x64), four 16-col tiles
        #pragma unroll
        for (int t = 0; t < 4; ++t) {
            FragB bV;
            const unsigned int* vp = (const unsigned int*)&Vt[w][t * 16 + l15][16 * h2];
            #pragma unroll
            for (int j = 0; j < 8; ++j) bV.u[j] = vp[j];
            O[t] = __builtin_amdgcn_wmma_f32_16x16x32_bf16(
                false, pf.v, false, bV.v, (short)0, O[t], false, false);
        }
        __syncthreads();
    }

    // normalize and store concatenated-head output
    #pragma unroll
    for (int r = 0; r < 8; ++r) {
        const int row = r + 8 * h2;
        float L = Lrow[w][row];
        float inv = (L > 0.0f) ? (1.0f / L) : 0.0f;
        const size_t orow = ((size_t)b * 2048 + q0 + row) * 768 + (size_t)h * 64;
        #pragma unroll
        for (int t = 0; t < 4; ++t)
            Obf[orow + t * 16 + l15] = f2bf(O[t][r] * inv);
    }
}

// ---------------- fused residual-add + LayerNorm (D=768) ----------------
__global__ __launch_bounds__(256)
void k_add_ln(const float* __restrict__ a, const float* __restrict__ bsrc,
              const float* __restrict__ g, const float* __restrict__ be,
              float* __restrict__ outF, unsigned short* __restrict__ outBf)
{
    __shared__ float red[256];
    __shared__ float stat[2];
    const int row = blockIdx.x;
    const int tid = threadIdx.x;
    const size_t base = (size_t)row * 768;
    float v[3];
    float s = 0.0f;
    #pragma unroll
    for (int j = 0; j < 3; ++j) {
        const int idx = tid + j * 256;
        v[j] = a[base + idx] + bsrc[base + idx];
        s += v[j];
    }
    red[tid] = s; __syncthreads();
    for (int off = 128; off > 0; off >>= 1) {
        if (tid < off) red[tid] += red[tid + off];
        __syncthreads();
    }
    if (tid == 0) stat[0] = red[0] * (1.0f / 768.0f);
    __syncthreads();
    const float mu = stat[0];
    float s2 = 0.0f;
    #pragma unroll
    for (int j = 0; j < 3; ++j) { float d = v[j] - mu; s2 += d * d; }
    red[tid] = s2; __syncthreads();
    for (int off = 128; off > 0; off >>= 1) {
        if (tid < off) red[tid] += red[tid + off];
        __syncthreads();
    }
    if (tid == 0) stat[1] = red[0] * (1.0f / 768.0f);
    __syncthreads();
    const float rstd = rsqrtf(stat[1] + 1e-5f);
    #pragma unroll
    for (int j = 0; j < 3; ++j) {
        const int idx = tid + j * 256;
        float o = (v[j] - mu) * rstd * g[idx] + be[idx];
        outF[base + idx] = o;
        if (outBf) outBf[base + idx] = f2bf(o);
    }
}

// ---------------- host orchestration ----------------
extern "C" void kernel_launch(void* const* d_in, const int* in_sizes, int n_in,
                              void* d_out, int out_size, void* d_ws, size_t ws_size,
                              hipStream_t stream)
{
    (void)in_sizes; (void)n_in; (void)out_size; (void)ws_size;
    const float* x   = (const float*)d_in[0];
    const float* Wq  = (const float*)d_in[1];
    const float* bq  = (const float*)d_in[2];
    const float* Wk  = (const float*)d_in[3];
    const float* bk  = (const float*)d_in[4];
    const float* Wv  = (const float*)d_in[5];
    const float* bv  = (const float*)d_in[6];
    const float* Wo  = (const float*)d_in[7];
    const float* bo  = (const float*)d_in[8];
    const float* W1  = (const float*)d_in[9];
    const float* b1  = (const float*)d_in[10];
    const float* W2  = (const float*)d_in[11];
    const float* b2  = (const float*)d_in[12];
    const float* g1  = (const float*)d_in[13];
    const float* be1 = (const float*)d_in[14];
    const float* g2  = (const float*)d_in[15];
    const float* be2 = (const float*)d_in[16];
    float* out = (float*)d_out;

    const int S = 2048, D = 768, F = 3072, M = 2 * S;  // B=2

    char* w = (char*)d_ws;
    size_t off = 0;
    auto alloc = [&](size_t bytes) -> void* {
        void* p = w + off;
        off = (off + bytes + 255) & ~(size_t)255;
        return p;
    };
    unsigned short* Xbf   = (unsigned short*)alloc((size_t)M * D * 2);
    unsigned short* Wqbf  = (unsigned short*)alloc((size_t)D * D * 2);
    unsigned short* Wkbf  = (unsigned short*)alloc((size_t)D * D * 2);
    unsigned short* Wvbf  = (unsigned short*)alloc((size_t)D * D * 2);
    unsigned short* Wobf  = (unsigned short*)alloc((size_t)D * D * 2);
    unsigned short* W1bf  = (unsigned short*)alloc((size_t)D * F * 2);
    unsigned short* W2bf  = (unsigned short*)alloc((size_t)F * D * 2);
    unsigned short* Qbf   = (unsigned short*)alloc((size_t)M * D * 2);
    unsigned short* Kbf   = (unsigned short*)alloc((size_t)M * D * 2);
    unsigned short* Vbf   = (unsigned short*)alloc((size_t)M * D * 2);
    unsigned short* Abf   = (unsigned short*)alloc((size_t)M * D * 2);  // attention out
    unsigned short* X2bf  = (unsigned short*)alloc((size_t)M * D * 2);
    unsigned short* Hbf   = (unsigned short*)alloc((size_t)M * F * 2);
    float*          TmpF  = (float*)alloc((size_t)M * D * 4);           // proj/ffn fp32 out
    float*          X2F   = (float*)alloc((size_t)M * D * 4);

    auto cvt = [&](const float* src, unsigned short* dst, size_t n) {
        int blocks = (int)((n + 255) / 256);
        k_f32_to_bf16<<<blocks, 256, 0, stream>>>(src, dst, (int)n);
    };
    cvt(x,  Xbf,  (size_t)M * D);
    cvt(Wq, Wqbf, (size_t)D * D);
    cvt(Wk, Wkbf, (size_t)D * D);
    cvt(Wv, Wvbf, (size_t)D * D);
    cvt(Wo, Wobf, (size_t)D * D);
    cvt(W1, W1bf, (size_t)D * F);
    cvt(W2, W2bf, (size_t)F * D);

    // QKV projections (MODE 2: head-split bf16 output)
    dim3 gDD(D / 128, M / 128);
    k_gemm_bf16<2, 0><<<gDD, 256, 0, stream>>>(Xbf, Wqbf, bq, Qbf, M, D, D);
    k_gemm_bf16<2, 0><<<gDD, 256, 0, stream>>>(Xbf, Wkbf, bk, Kbf, M, D, D);
    k_gemm_bf16<2, 0><<<gDD, 256, 0, stream>>>(Xbf, Wvbf, bv, Vbf, M, D, D);

    // flash attention: 3072 waves, 4 per block
    k_attention<<<768, 128, 0, stream>>>(Qbf, Kbf, Vbf, Abf);

    // output projection (fp32 out)
    k_gemm_bf16<1, 0><<<gDD, 256, 0, stream>>>(Abf, Wobf, bo, TmpF, M, D, D);

    // x2 = LN(x + attn_proj)
    k_add_ln<<<M, 256, 0, stream>>>(x, TmpF, g1, be1, X2F, X2bf);

    // FFN: H = relu(x2 @ W1 + b1) ; ff = H @ W2 + b2
    dim3 gDF(F / 128, M / 128);
    k_gemm_bf16<0, 1><<<gDF, 256, 0, stream>>>(X2bf, W1bf, b1, Hbf, M, F, D);
    k_gemm_bf16<1, 0><<<gDD, 256, 0, stream>>>(Hbf, W2bf, b2, TmpF, M, D, F);

    // out = LN(x2 + ff)
    k_add_ln<<<M, 256, 0, stream>>>(X2F, TmpF, g2, be2, out, (unsigned short*)nullptr);
}
